// PrimeKGEncoder_8443905704465
// MI455X (gfx1250) — matrix operand order repaired and checked by reading, not compiled, not run on hardware
//
#include <hip/hip_runtime.h>
#include <hip/hip_bf16.h>

// ---------------------------------------------------------------------------
// Types for CDNA5 WMMA (gfx1250, wave32)
// ---------------------------------------------------------------------------
typedef __bf16 bf16_t;
typedef __attribute__((ext_vector_type(16))) __bf16 v16bf;
typedef __attribute__((ext_vector_type(8)))  __bf16 v8bf;
typedef __attribute__((ext_vector_type(8)))  float  v8f;

// ---------------------------------------------------------------------------
// Elementwise helpers
// ---------------------------------------------------------------------------
__global__ void fill_zero_kernel(float* __restrict__ p, long long n) {
    long long t = (long long)blockIdx.x * blockDim.x + threadIdx.x;
    if (t < n) p[t] = 0.0f;
}

// x = s ; xb = (bf16)s     (initial embedding load)
__global__ void copy_cvt_kernel(const float* __restrict__ s, float* __restrict__ x,
                                bf16_t* __restrict__ xb, long long n) {
    long long t = (long long)blockIdx.x * blockDim.x + threadIdx.x;
    if (t < n) { float v = s[t]; x[t] = v; xb[t] = (bf16_t)v; }
}

// x = relu(x + add) ; xb = (bf16)x      (layer epilogue, fused convert)
__global__ void resrelu_cvt_kernel(float* __restrict__ x, const float* __restrict__ add,
                                   bf16_t* __restrict__ xb, long long n) {
    long long t = (long long)blockIdx.x * blockDim.x + threadIdx.x;
    if (t < n) { float v = fmaxf(x[t] + add[t], 0.0f); x[t] = v; xb[t] = (bf16_t)v; }
}

// aggb = (bf16)(agg / max(cnt,1))       (mean + convert, fused)
__global__ void div_cvt_kernel(const float* __restrict__ agg, const float* __restrict__ cnt,
                               bf16_t* __restrict__ aggb, long long n) {
    long long t = (long long)blockIdx.x * blockDim.x + threadIdx.x;
    if (t < n) aggb[t] = (bf16_t)(agg[t] / fmaxf(cnt[t >> 8], 1.0f));
}

// Wt[n*K + k] = (bf16) W[k*ncols + n]   (transpose + convert; coalesced writes)
__global__ void wtrans_kernel(const float* __restrict__ W, bf16_t* __restrict__ Wt,
                              int K, int ncols) {
    int t = blockIdx.x * blockDim.x + threadIdx.x;
    if (t >= K * ncols) return;
    int n = t / K;
    int k = t - n * K;
    Wt[t] = (bf16_t)W[k * ncols + n];
}

// ---------------------------------------------------------------------------
// Graph aggregation (mean): atomic scatter-add of features + edge counts
// ---------------------------------------------------------------------------
__global__ void count_kernel(const int* __restrict__ dst, float* __restrict__ cnt, int E) {
    int e = blockIdx.x * blockDim.x + threadIdx.x;
    if (e < E) atomicAdd(cnt + dst[e], 1.0f);
}

// one float4 chunk per thread: 64 threads cover one edge's 256 features
__global__ void scatter_kernel(const float* __restrict__ x, const int* __restrict__ src,
                               const int* __restrict__ dst, float* __restrict__ agg, int E) {
    long long t = (long long)blockIdx.x * blockDim.x + threadIdx.x;
    long long total = (long long)E << 6;       // E * 64
    if (t >= total) return;
    int e   = (int)(t >> 6);
    int seg = (int)(t & 63);
    int s = src[e], d = dst[e];
    const float4 v = ((const float4*)(x + (long long)s * 256))[seg];
    float* o = agg + (long long)d * 256 + (long long)seg * 4;
    atomicAdd(o + 0, v.x);
    atomicAdd(o + 1, v.y);
    atomicAdd(o + 2, v.z);
    atomicAdd(o + 3, v.w);
}

// ---------------------------------------------------------------------------
// WMMA GEMM, 2x2 register-tiled: each wave computes a 32x32 C tile.
// C[rows x ncols](f32)  (op)=  A[rows x K](bf16) * W  (+ bias)
// W supplied transposed: Wt[ncols x K](bf16 row-major), Wt[n][k] = W[k][n].
// Per K-step (32): 2 A fragments + 2 B fragments -> 4 v_wmma_f32_16x16x32_bf16
// (2 b128 loads per WMMA instead of 4 for the untiled version).
//
// A-fragment lane layout (ISA 7.12.2, 16-bit A 16x32): lane l (m=l&15):
//   lanes 0-15 : halves 0..7 = K k0+0..7,  halves 8..15 = K k0+16..23
//   lanes 16-31: halves 0..7 = K k0+8..15, halves 8..15 = K k0+24..31
// B-fragment (32x16): lane l holds column n=l&15; same K order as A,
//   -> two 16B loads from Wt row n (k0+akb, k0+16+akb).
// C/D layout: lane l col n=l&15; VGPR j -> row j (+8 for lanes 16-31).
//
// mode: 0 => C += acc+bias ; 1 => C = acc+bias ; 2 => C = relu(acc+bias) ;
//       3 => Cb = (bf16)relu(acc+bias)   (bf16 output, f32 C unused)
// ---------------------------------------------------------------------------
__device__ __forceinline__ void store_tile(const v8f& acc, float* __restrict__ C,
                                           bf16_t* __restrict__ Cb, int row0, int col,
                                           int ncols, float bv, int mode) {
#pragma unroll
    for (int j = 0; j < 8; ++j) {
        float v = acc[j] + bv;
        long long off = (long long)(row0 + j) * ncols + col;
        if (mode == 0)      C[off] += v;
        else if (mode == 1) C[off] = v;
        else if (mode == 2) C[off] = fmaxf(v, 0.0f);
        else                Cb[off] = (bf16_t)fmaxf(v, 0.0f);
    }
}

__global__ __launch_bounds__(32)
void wmma_gemm_bf16(const bf16_t* __restrict__ A, const bf16_t* __restrict__ Wt,
                    const float* __restrict__ bias, float* __restrict__ C,
                    bf16_t* __restrict__ Cb, int K, int ncols, int mode) {
    const int r0       = blockIdx.x * 32;
    const int c0       = blockIdx.y * 32;
    const int lane     = threadIdx.x;        // 0..31
    const int half_sel = lane >> 4;          // 0 | 1
    const int mn       = lane & 15;
    const long long arow0 = (long long)(r0 + mn) * K;
    const long long arow1 = arow0 + (long long)16 * K;
    const long long brow0 = (long long)(c0 + mn) * K;
    const long long brow1 = brow0 + (long long)16 * K;
    const int akb = half_sel * 8;
    const int bkb = half_sel * 16;

    v8f acc00 = {}, acc01 = {}, acc10 = {}, acc11 = {};
    for (int k0 = 0; k0 < K; k0 += 32) {
        __builtin_prefetch(A + arow0 + k0 + 64, 0, 1);   // next K tile of A
        union { v16bf v; v8bf h[2]; } a0, a1;
        a0.h[0] = *(const v8bf*)(A + arow0 + k0 + akb);        // 16B aligned
        a0.h[1] = *(const v8bf*)(A + arow0 + k0 + 16 + akb);
        a1.h[0] = *(const v8bf*)(A + arow1 + k0 + akb);
        a1.h[1] = *(const v8bf*)(A + arow1 + k0 + 16 + akb);
        union { v16bf v; v8bf h[2]; } b0u, b1u;
        b0u.h[0] = *(const v8bf*)(Wt + brow0 + k0 + akb);      // 16B aligned
        b0u.h[1] = *(const v8bf*)(Wt + brow0 + k0 + 16 + akb);
        b1u.h[0] = *(const v8bf*)(Wt + brow1 + k0 + akb);
        b1u.h[1] = *(const v8bf*)(Wt + brow1 + k0 + 16 + akb);
        v16bf b0 = b0u.v;
        v16bf b1 = b1u.v;
        acc00 = __builtin_amdgcn_wmma_f32_16x16x32_bf16(false, a0.v, false, b0, (short)0, acc00, false, false);
        acc01 = __builtin_amdgcn_wmma_f32_16x16x32_bf16(false, a0.v, false, b1, (short)0, acc01, false, false);
        acc10 = __builtin_amdgcn_wmma_f32_16x16x32_bf16(false, a1.v, false, b0, (short)0, acc10, false, false);
        acc11 = __builtin_amdgcn_wmma_f32_16x16x32_bf16(false, a1.v, false, b1, (short)0, acc11, false, false);
    }

    const int col0 = c0 + mn;
    const int col1 = col0 + 16;
    const float bv0 = bias ? bias[col0] : 0.0f;
    const float bv1 = bias ? bias[col1] : 0.0f;
    const int row0 = r0 + half_sel * 8;
    store_tile(acc00, C, Cb, row0,      col0, ncols, bv0, mode);
    store_tile(acc01, C, Cb, row0,      col1, ncols, bv1, mode);
    store_tile(acc10, C, Cb, row0 + 16, col0, ncols, bv0, mode);
    store_tile(acc11, C, Cb, row0 + 16, col1, ncols, bv1, mode);
}

// ---------------------------------------------------------------------------
// LayerNorm over 128 cols: one wave32 per row, float4 per lane
// ---------------------------------------------------------------------------
__global__ __launch_bounds__(128)
void ln_kernel(const float* __restrict__ y, const float* __restrict__ g,
               const float* __restrict__ be, float* __restrict__ out, int n) {
    int row  = blockIdx.x * 4 + (threadIdx.x >> 5);
    int lane = threadIdx.x & 31;
    if (row >= n) return;
    float4 v = ((const float4*)(y + (long long)row * 128))[lane];
    float s = v.x + v.y + v.z + v.w;
    for (int o = 16; o > 0; o >>= 1) s += __shfl_xor(s, o, 32);
    float mu = s * (1.0f / 128.0f);
    float dx = v.x - mu, dy = v.y - mu, dz = v.z - mu, dw = v.w - mu;
    float q = dx * dx + dy * dy + dz * dz + dw * dw;
    for (int o = 16; o > 0; o >>= 1) q += __shfl_xor(q, o, 32);
    float inv = rsqrtf(q * (1.0f / 128.0f) + 1e-5f);
    float4 go = ((const float4*)g)[lane];
    float4 bo = ((const float4*)be)[lane];
    float4 r;
    r.x = dx * inv * go.x + bo.x;
    r.y = dy * inv * go.y + bo.y;
    r.z = dz * inv * go.z + bo.z;
    r.w = dw * inv * go.w + bo.w;
    ((float4*)(out + (long long)row * 128))[lane] = r;
}

// ---------------------------------------------------------------------------
// Host orchestration
// ---------------------------------------------------------------------------
static inline unsigned gblk(long long n, int b) { return (unsigned)((n + b - 1) / b); }

extern "C" void kernel_launch(void* const* d_in, const int* in_sizes, int n_in,
                              void* d_out, int out_size, void* d_ws, size_t ws_size,
                              hipStream_t stream) {
    (void)n_in; (void)out_size; (void)ws_size;
    const int H = 256, OUTC = 128, LAYERS = 3;
    const int NN[3] = { 8000, 20000, 16000 };      // drug, protein, effect
    const int NMAX = 20000;

    // -------- input pointers (setup_inputs dict order) --------
    const float* emb[3] = { (const float*)d_in[0], (const float*)d_in[1], (const float*)d_in[2] };
    auto Wl = [&](int l, int r) { return (const float*)d_in[3 + (l * 4 + r) * 3 + 0]; };
    auto bl = [&](int l, int r) { return (const float*)d_in[3 + (l * 4 + r) * 3 + 1]; };
    auto Wr = [&](int l, int r) { return (const float*)d_in[3 + (l * 4 + r) * 3 + 2]; };
    auto pj = [&](int t, int k) { return (const float*)d_in[39 + t * 6 + k]; };
    const int* ei[4]; int Ecnt[4];
    for (int r = 0; r < 4; ++r) { ei[r] = (const int*)d_in[57 + r]; Ecnt[r] = in_sizes[57 + r] / 2; }
    const int rsrc[4] = { 0, 0, 1, 0 };            // drug_protein, drug_drug, protein_protein, drug_effect
    const int rdst[4] = { 1, 0, 1, 2 };

    // -------- workspace carve --------
    char* w = (char*)d_ws;
    auto carve = [&](size_t bytes) { char* p = w; w += (bytes + 255) & ~(size_t)255; return p; };
    float*  x[3];  for (int t = 0; t < 3; ++t) x[t]  = (float*) carve((size_t)NN[t] * H * 4);
    float*  nw[3]; for (int t = 0; t < 3; ++t) nw[t] = (float*) carve((size_t)NN[t] * H * 4);
    bf16_t* xb[3]; for (int t = 0; t < 3; ++t) xb[t] = (bf16_t*)carve((size_t)NN[t] * H * 2);
    float*  agg  = (float*) carve((size_t)NMAX * H * 4);
    bf16_t* aggb = (bf16_t*)carve((size_t)NMAX * H * 2);
    float*  cnt  = (float*) carve((size_t)NMAX * 4);
    bf16_t* Wt   = (bf16_t*)carve((size_t)H * H * 2);
    float*  ytmp = (float*) carve((size_t)NMAX * OUTC * 4);
    bf16_t* hb   = aggb;           // proj-phase alias: [NMAX x 256] bf16

    // -------- init x / xb from embeddings --------
    for (int t = 0; t < 3; ++t) {
        long long n = (long long)NN[t] * H;
        copy_cvt_kernel<<<gblk(n, 256), 256, 0, stream>>>(emb[t], x[t], xb[t], n);
    }

    // -------- message-passing layers --------
    for (int l = 0; l < LAYERS; ++l) {
        for (int t = 0; t < 3; ++t) {
            long long n = (long long)NN[t] * H;
            fill_zero_kernel<<<gblk(n, 256), 256, 0, stream>>>(nw[t], n);
        }
        for (int r = 0; r < 4; ++r) {
            const int st = rsrc[r], dt = rdst[r];
            const int nd = NN[dt], E = Ecnt[r];
            const long long na = (long long)nd * H;
            fill_zero_kernel<<<gblk(na, 256), 256, 0, stream>>>(agg, na);
            fill_zero_kernel<<<gblk(nd, 256), 256, 0, stream>>>(cnt, nd);
            count_kernel  <<<gblk(E, 256), 256, 0, stream>>>(ei[r] + E, cnt, E);
            scatter_kernel<<<gblk((long long)E * 64, 256), 256, 0, stream>>>(x[st], ei[r], ei[r] + E, agg, E);
            div_cvt_kernel<<<gblk(na, 256), 256, 0, stream>>>(agg, cnt, aggb, na);
            // neighbor branch: nw[dt] += mean_agg @ Wl + bl
            wtrans_kernel<<<gblk(H * H, 256), 256, 0, stream>>>(Wl(l, r), Wt, H, H);
            wmma_gemm_bf16<<<dim3(nd / 32, H / 32), 32, 0, stream>>>(aggb, Wt, bl(l, r), nw[dt], nullptr, H, H, 0);
            // root branch: nw[dt] += x[dt] @ Wr
            wtrans_kernel<<<gblk(H * H, 256), 256, 0, stream>>>(Wr(l, r), Wt, H, H);
            wmma_gemm_bf16<<<dim3(nd / 32, H / 32), 32, 0, stream>>>(xb[dt], Wt, nullptr, nw[dt], nullptr, H, H, 0);
        }
        for (int t = 0; t < 3; ++t) {
            long long n = (long long)NN[t] * H;
            resrelu_cvt_kernel<<<gblk(n, 256), 256, 0, stream>>>(x[t], nw[t], xb[t], n);
        }
    }

    // -------- output projection + LayerNorm --------
    long long ooff = 0;
    for (int t = 0; t < 3; ++t) {
        const int n = NN[t];
        // h(bf16) = relu(x @ W1 + b1)    (bf16 written directly by GEMM, mode 3)
        wtrans_kernel<<<gblk(H * H, 256), 256, 0, stream>>>(pj(t, 0), Wt, H, H);
        wmma_gemm_bf16<<<dim3(n / 32, H / 32), 32, 0, stream>>>(xb[t], Wt, pj(t, 1), nullptr, hb, H, H, 3);
        // y = h @ W2 + b2
        wtrans_kernel<<<gblk(H * OUTC, 256), 256, 0, stream>>>(pj(t, 2), Wt, H, OUTC);
        wmma_gemm_bf16<<<dim3(n / 32, OUTC / 32), 32, 0, stream>>>(hb, Wt, pj(t, 3), ytmp, nullptr, H, OUTC, 1);
        // LayerNorm -> d_out segment
        ln_kernel<<<gblk(n, 4), 128, 0, stream>>>(ytmp, pj(t, 4), pj(t, 5), (float*)d_out + ooff, n);
        ooff += (long long)n * OUTC;
    }
}
